// SelfAttention_33036888441134
// MI455X (gfx1250) — matrix operand order repaired and checked by reading, not compiled, not run on hardware
//
#include <hip/hip_runtime.h>
#include <hip/hip_bf16.h>

// ---------------------------------------------------------------------------
// Self-attention for MI455X (gfx1250): bf16 WMMA + TDM async tensor loads.
// B=4, S=2048, E=1024, H=16, D=64.
// ---------------------------------------------------------------------------

#define EDIM 1024
#define SDIM 2048
#define HDIM 16
#define DDIM 64
#define BDIM 4
#define MROWS (BDIM * SDIM)   // 8192
#define KSTEP 64              // K-slice per TDM stage

typedef __bf16 v8bf  __attribute__((ext_vector_type(8)));
typedef __bf16 v16bf __attribute__((ext_vector_type(16)));
typedef float  v8f   __attribute__((ext_vector_type(8)));
typedef unsigned int v4u __attribute__((ext_vector_type(4)));
typedef int v4i __attribute__((ext_vector_type(4)));
typedef int v8i __attribute__((ext_vector_type(8)));

__device__ __forceinline__ v16bf cat8(v8bf lo, v8bf hi) {
    return __builtin_shufflevector(lo, hi, 0, 1, 2, 3, 4, 5, 6, 7,
                                   8, 9, 10, 11, 12, 13, 14, 15);
}

__device__ __forceinline__ v8f wmma_bf16(v16bf a, v16bf b, v8f c) {
    // D = A(16x32 bf16) * B(32x16 bf16) + C(16x16 f32)
    return __builtin_amdgcn_wmma_f32_16x16x32_bf16(
        /*neg_a=*/false, a, /*neg_b=*/false, b,
        /*c_mod=*/(short)0, c, /*reuse_a=*/false, /*reuse_b=*/false);
}

// ------------------------- TDM descriptor helpers --------------------------
// D# per CDNA5 ISA ch.8.  Group 1 (dims/strides) is loop-invariant; build it
// once.  Group 0 carries the per-call LDS and global addresses.

__device__ __forceinline__ v8i tdm_make_g1(
    unsigned tile_k, unsigned tile_rows, unsigned row_stride_elems) {
    v8i g1;
    g1[0] = (int)(1u << 16);                      // data_size=1 (2 bytes/elem)
    g1[1] = (int)((tile_k & 0xffffu) << 16);      // tensor_dim0[15:0]
    g1[2] = (int)((tile_k >> 16) |                // tensor_dim0[31:16]
                  ((tile_rows & 0xffffu) << 16)); // tensor_dim1[15:0]
    g1[3] = (int)((tile_rows >> 16) |             // tensor_dim1[31:16]
                  (tile_k << 16));                // tile_dim0 = tile_k
    g1[4] = (int)(tile_rows & 0xffffu);           // tile_dim1 (tile_dim2=0)
    g1[5] = (int)row_stride_elems;                // tensor_dim0_stride[31:0]
    g1[6] = 0;                                    // stride0 hi / stride1 lo
    g1[7] = 0;
    return g1;
}

__device__ __forceinline__ void tdm_issue(
    const __bf16* gsrc, unsigned lds_off, v8i g1) {
    unsigned long long ga = (unsigned long long)(uintptr_t)gsrc;
    v4u g0;
    g0[0] = 1u;                                   // count=1, user descriptor
    g0[1] = lds_off;                              // lds_addr (bytes)
    g0[2] = (unsigned)(ga & 0xffffffffu);         // global_addr[31:0]
    g0[3] = (unsigned)((ga >> 32) & 0x1ffffffu)   // global_addr[56:32]
            | (2u << 30);                         // type=2 ("image")
    v4i z4 = {0, 0, 0, 0};                        // 2D: groups 2/3 unused
    v8i z8 = {0, 0, 0, 0, 0, 0, 0, 0};
    __builtin_amdgcn_tensor_load_to_lds(g0, g1, z4, z4, z8, 0);
}

// --------------------------- elementwise casts -----------------------------

__global__ __launch_bounds__(256) void cast_f32_to_bf16(
    const float* __restrict__ in, __bf16* __restrict__ out, int n) {
    int i = blockIdx.x * blockDim.x + threadIdx.x;
    if (i < n) out[i] = (__bf16)in[i];
}

// out[n][k] = (bf16) in[k][n]   (1024x1024)
__global__ __launch_bounds__(256) void transpose_cast_w(
    const float* __restrict__ in, __bf16* __restrict__ out) {
    int idx = blockIdx.x * blockDim.x + threadIdx.x;  // 0 .. 1048575
    int nn = idx >> 10;
    int kk = idx & 1023;
    out[idx] = (__bf16)in[kk * EDIM + nn];
}

// ------------------------------- GEMM --------------------------------------
// C[M x 1024] = A_bf16[M x 1024] * W[1024 x 1024] + bias, W given as Wt
// (row-major N x K, bf16).  8 waves/block; block tile 32(M) x 256(N); each
// wave computes 16 x 64.  A and B K=64 slices staged in LDS by the Tensor
// Data Mover (wave 0 issues, TENSORcnt + barrier), double-buffered.
//
// mode 0: out = bf16 (B,H,S,D)   (Q / K)
// mode 1: out = bf16 (B,H,D,S)   (V transposed)
// mode 2: out = f32  (B*S, E)    (final projection)
__global__ __launch_bounds__(256) void gemm_bf16(
    const __bf16* __restrict__ A, const __bf16* __restrict__ Bt,
    const float* __restrict__ bias, void* __restrict__ out, int mode) {
    __shared__ __bf16 lds_b[2][256 * KSTEP];   // 2 x 32 KB
    __shared__ __bf16 lds_a[2][32 * KSTEP];    // 2 x 4 KB

    const int lane = threadIdx.x & 31;
    const int wave = threadIdx.x >> 5;
    const int half = lane >> 4;
    const int l16  = lane & 15;
    const int mbase0 = blockIdx.x * 32;
    const int nbase0 = blockIdx.y * 256;
    const int mbase  = mbase0 + (wave & 1) * 16;
    const int nloc   = (wave >> 1) * 64;          // wave's n offset in tile
    const int arow   = (wave & 1) * 16 + l16;     // wave's A row in LDS tile

    const unsigned lb0 = (unsigned)(uintptr_t)&lds_b[0][0];
    const unsigned lb1 = (unsigned)(uintptr_t)&lds_b[1][0];
    const unsigned la0 = (unsigned)(uintptr_t)&lds_a[0][0];
    const unsigned la1 = (unsigned)(uintptr_t)&lds_a[1][0];

    const v8i g1b = tdm_make_g1(KSTEP, 256, EDIM);
    const v8i g1a = tdm_make_g1(KSTEP, 32, EDIM);

    v8f acc[4] = {v8f{}, v8f{}, v8f{}, v8f{}};

    // prologue: wave 0 DMAs k-slice 0 into buffer 0
    if (wave == 0) {
        tdm_issue(Bt + (size_t)nbase0 * EDIM, lb0, g1b);
        tdm_issue(A + (size_t)mbase0 * EDIM, la0, g1a);
        __builtin_amdgcn_s_wait_tensorcnt(0);
    }
    __syncthreads();

    for (int kb = 0; kb < EDIM; kb += KSTEP) {
        const int buf = (kb / KSTEP) & 1;
        const bool more = (kb + KSTEP) < EDIM;
        if (wave == 0 && more) {   // DMA next k-slice into the other buffer
            tdm_issue(Bt + (size_t)nbase0 * EDIM + kb + KSTEP,
                      buf ? lb0 : lb1, g1b);
            tdm_issue(A + (size_t)mbase0 * EDIM + kb + KSTEP,
                      buf ? la0 : la1, g1a);
        }

#pragma unroll
        for (int kf = 0; kf < 2; ++kf) {
            // A fragment: lane row arow, chunks kf*32 + {16*half, 16*half+8}
            const __bf16* abase = &lds_a[buf][arow * KSTEP + kf * 32];
            v8bf alo = *(const v8bf*)(abase + 16 * half);
            v8bf ahi = *(const v8bf*)(abase + 16 * half + 8);
            v16bf af = cat8(alo, ahi);
            // batch the 4 B-fragment LDS loads, then issue 4 WMMAs
            v16bf bfr[4];
#pragma unroll
            for (int nf = 0; nf < 4; ++nf)
                bfr[nf] = *(const v16bf*)
                    &lds_b[buf][(nloc + nf * 16 + l16) * KSTEP +
                                kf * 32 + 16 * half];
#pragma unroll
            for (int nf = 0; nf < 4; ++nf)
                acc[nf] = wmma_bf16(af, bfr[nf], acc[nf]);
        }

        if (wave == 0 && more) __builtin_amdgcn_s_wait_tensorcnt(0);
        __syncthreads();
    }

#pragma unroll
    for (int nf = 0; nf < 4; ++nf) {
        int n = nbase0 + nloc + nf * 16 + l16;
        float bv = bias[n];
#pragma unroll
        for (int r = 0; r < 8; ++r) {
            float v = acc[nf][r] + bv;
            int m = mbase + r + 8 * half;     // row of (B*S, E) matrix
            if (mode == 2) {
                ((float*)out)[(size_t)m * EDIM + n] = v;
            } else {
                int b = m >> 11;              // m / SDIM
                int s = m & (SDIM - 1);
                int h = n >> 6;               // n / DDIM
                int d = n & (DDIM - 1);
                __bf16 bb = (__bf16)v;
                if (mode == 0) {  // (B,H,S,D)
                    ((__bf16*)out)[(((size_t)b * HDIM + h) * SDIM + s) * DDIM + d] = bb;
                } else {          // (B,H,D,S)
                    ((__bf16*)out)[(((size_t)b * HDIM + h) * DDIM + d) * SDIM + s] = bb;
                }
            }
        }
    }
}

// --------------------------- flash attention -------------------------------
// One wave per (b, h, 16-query-row tile). Online softmax over 64-key tiles.
// Q,K in (B,H,S,D) bf16; V in (B,H,D,S) bf16. Writes bf16 ctx (B*S, E).
__global__ __launch_bounds__(256) void attn_flash(
    const __bf16* __restrict__ Q, const __bf16* __restrict__ K,
    const __bf16* __restrict__ Vt, __bf16* __restrict__ ctxout) {
    __shared__ __bf16 lds[8][16 * 64];  // per-wave 2KB P-transpose tile

    const int lane = threadIdx.x & 31;
    const int wave = threadIdx.x >> 5;
    const int half = lane >> 4;
    const int l16  = lane & 15;

    int t  = blockIdx.x * 8 + wave;     // 0 .. 8191 tiles
    int ts = t & 127;                   // query-tile index within head
    int h  = (t >> 7) & (HDIM - 1);
    int b  = t >> 11;

    const size_t headoff = ((size_t)b * HDIM + h) * SDIM * DDIM;
    const __bf16* q  = Q  + headoff;    // (S, D)
    const __bf16* kk = K  + headoff;    // (S, D)
    const __bf16* vt = Vt + headoff;    // (D, S)

    // Q A-fragments: 16 rows x 64 (two K=32 fragments)
    int qrow = ts * 16 + l16;
    v16bf qa[2];
#pragma unroll
    for (int kf = 0; kf < 2; ++kf) {
        v8bf lo = *(const v8bf*)(q + (size_t)qrow * DDIM + kf * 32 + 16 * half);
        v8bf hi = *(const v8bf*)(q + (size_t)qrow * DDIM + kf * 32 + 16 * half + 8);
        qa[kf] = cat8(lo, hi);
    }

    v8f ctx[4] = {v8f{}, v8f{}, v8f{}, v8f{}};
    float mrun[8], lrun[8];
#pragma unroll
    for (int r = 0; r < 8; ++r) { mrun[r] = -3.0e38f; lrun[r] = 0.0f; }

    const float scale = 0.125f;  // 1/sqrt(64)

    for (int kt = 0; kt < SDIM; kt += 64) {
        // scores S = Q * K^T  (16 x 64), 4 C-fragments
        v8f sc[4];
#pragma unroll
        for (int nf = 0; nf < 4; ++nf) {
            int key = kt + nf * 16 + l16;
            v16bf b0 = *(const v16bf*)(kk + (size_t)key * DDIM + 16 * half);
            v16bf b1 = *(const v16bf*)(kk + (size_t)key * DDIM + 32 + 16 * half);
            if (kt + 64 < SDIM)  // prefetch next key tile
                __builtin_prefetch(kk + (size_t)(key + 64) * DDIM, 0, 1);
            v8f sf = v8f{};
            sf = wmma_bf16(qa[0], b0, sf);
            sf = wmma_bf16(qa[1], b1, sf);
#pragma unroll
            for (int r = 0; r < 8; ++r) sf[r] *= scale;
            sc[nf] = sf;
        }

        // row max (row = r + 8*half; columns spread over 16-lane halves)
        float tmax[8];
#pragma unroll
        for (int r = 0; r < 8; ++r) {
            float m0 = fmaxf(fmaxf(sc[0][r], sc[1][r]), fmaxf(sc[2][r], sc[3][r]));
#pragma unroll
            for (int msk = 1; msk < 16; msk <<= 1)
                m0 = fmaxf(m0, __shfl_xor(m0, msk, 32));
            tmax[r] = m0;
        }

        float alpha[8];
#pragma unroll
        for (int r = 0; r < 8; ++r) {
            float nm = fmaxf(mrun[r], tmax[r]);
            alpha[r] = __expf(mrun[r] - nm);
            mrun[r]  = nm;
        }

        // P = exp(S - m); write bf16 P to LDS in (row, col) order; row sums
        float psum[8];
#pragma unroll
        for (int r = 0; r < 8; ++r) psum[r] = 0.0f;
#pragma unroll
        for (int nf = 0; nf < 4; ++nf) {
#pragma unroll
            for (int r = 0; r < 8; ++r) {
                float p = __expf(sc[nf][r] - mrun[r]);
                psum[r] += p;
                lds[wave][(r + 8 * half) * 64 + nf * 16 + l16] = (__bf16)p;
            }
        }
#pragma unroll
        for (int r = 0; r < 8; ++r) {
#pragma unroll
            for (int msk = 1; msk < 16; msk <<= 1)
                psum[r] += __shfl_xor(psum[r], msk, 32);
            lrun[r] = alpha[r] * lrun[r] + psum[r];
        }

        // rescale ctx accumulators
#pragma unroll
        for (int nf = 0; nf < 4; ++nf)
#pragma unroll
            for (int r = 0; r < 8; ++r) ctx[nf][r] *= alpha[r];

        // reload P from LDS as A-fragments (16 x 64 -> two K=32 fragments)
        v16bf pa[2];
#pragma unroll
        for (int kf = 0; kf < 2; ++kf) {
            const __bf16* base = &lds[wave][l16 * 64 + kf * 32];
            v8bf lo = *(const v8bf*)(base + 16 * half);
            v8bf hi = *(const v8bf*)(base + 16 * half + 8);
            pa[kf] = cat8(lo, hi);
        }

        // ctx += P * V   (V fragments contiguous thanks to (D,S) layout)
#pragma unroll
        for (int nf = 0; nf < 4; ++nf) {
            int d = nf * 16 + l16;
            v16bf v0 = *(const v16bf*)(vt + (size_t)d * SDIM + kt + 16 * half);
            v16bf v1 = *(const v16bf*)(vt + (size_t)d * SDIM + kt + 32 + 16 * half);
            if (kt + 64 < SDIM)
                __builtin_prefetch(vt + (size_t)d * SDIM + kt + 64, 0, 1);
            ctx[nf] = wmma_bf16(pa[0], v0, ctx[nf]);
            ctx[nf] = wmma_bf16(pa[1], v1, ctx[nf]);
        }
    }

    // normalize and write ctx as bf16 (B*S, E), col = h*64 + d
#pragma unroll
    for (int r = 0; r < 8; ++r) {
        float inv = 1.0f / lrun[r];
        int srow = ts * 16 + r + 8 * half;
        size_t rowg = (size_t)b * SDIM + srow;
#pragma unroll
        for (int nf = 0; nf < 4; ++nf) {
            int col = h * DDIM + nf * 16 + l16;
            ctxout[rowg * EDIM + col] = (__bf16)(ctx[nf][r] * inv);
        }
    }
}

// ------------------------------- launch ------------------------------------

extern "C" void kernel_launch(void* const* d_in, const int* in_sizes, int n_in,
                              void* d_out, int out_size, void* d_ws, size_t ws_size,
                              hipStream_t stream) {
    (void)in_sizes; (void)n_in; (void)out_size; (void)ws_size;
    const float* x  = (const float*)d_in[0];
    const float* Wq = (const float*)d_in[1];
    const float* bq = (const float*)d_in[2];
    const float* Wk = (const float*)d_in[3];
    const float* bk = (const float*)d_in[4];
    const float* Wv = (const float*)d_in[5];
    const float* bv = (const float*)d_in[6];
    const float* Wo = (const float*)d_in[7];
    const float* bo = (const float*)d_in[8];

    // workspace layout (bf16 buffers)
    char* ws = (char*)d_ws;
    const size_t XB = (size_t)MROWS * EDIM * 2;      // 16 MB
    const size_t WB = (size_t)EDIM * EDIM * 2;       // 2 MB
    __bf16* xbf  = (__bf16*)(ws);
    __bf16* wqT  = (__bf16*)(ws + XB);
    __bf16* wkT  = (__bf16*)(ws + XB + WB);
    __bf16* wvT  = (__bf16*)(ws + XB + 2 * WB);
    __bf16* woT  = (__bf16*)(ws + XB + 3 * WB);
    __bf16* qb   = (__bf16*)(ws + XB + 4 * WB);
    __bf16* kb   = (__bf16*)(ws + 2 * XB + 4 * WB);
    __bf16* vtb  = (__bf16*)(ws + 3 * XB + 4 * WB);
    __bf16* ctxb = (__bf16*)(ws + 4 * XB + 4 * WB);

    // 1) casts
    {
        int n = MROWS * EDIM;
        cast_f32_to_bf16<<<(n + 255) / 256, 256, 0, stream>>>(x, xbf, n);
        int nw = EDIM * EDIM;
        transpose_cast_w<<<nw / 256, 256, 0, stream>>>(Wq, wqT);
        transpose_cast_w<<<nw / 256, 256, 0, stream>>>(Wk, wkT);
        transpose_cast_w<<<nw / 256, 256, 0, stream>>>(Wv, wvT);
        transpose_cast_w<<<nw / 256, 256, 0, stream>>>(Wo, woT);
    }

    // 2) projections
    dim3 ggrid(MROWS / 32, EDIM / 256);
    gemm_bf16<<<ggrid, 256, 0, stream>>>(xbf, wqT, bq, (void*)qb,  0);
    gemm_bf16<<<ggrid, 256, 0, stream>>>(xbf, wkT, bk, (void*)kb,  0);
    gemm_bf16<<<ggrid, 256, 0, stream>>>(xbf, wvT, bv, (void*)vtb, 1);

    // 3) attention: 8192 row-tiles, 8 waves per block
    attn_flash<<<8192 / 8, 256, 0, stream>>>(qb, kb, vtb, ctxb);

    // 4) output projection -> f32 d_out
    gemm_bf16<<<ggrid, 256, 0, stream>>>(ctxb, woT, bo, d_out, 2);
}
